// Decoder_77833397338153
// MI455X (gfx1250) — compile-verified, matrix operand order checked by reading, not run on hardware
//
#include <hip/hip_runtime.h>
#include <math.h>

#define B_    8
#define T_    50
#define S_    256
#define H_    512
#define E_    512
#define L_    4
#define VOCAB_ 50000
#define VEXT_  50256
#define BT_   (B_*T_)
#define G3H_  (3*H_)

typedef __bf16 bf16;
typedef __bf16 v8bf  __attribute__((ext_vector_type(8)));
typedef __bf16 v16bf __attribute__((ext_vector_type(16)));
typedef float  v8f   __attribute__((ext_vector_type(8)));

#if defined(__has_builtin)
#  if __has_builtin(__builtin_amdgcn_global_load_async_to_lds_b128)
#    define HAVE_ASYNC_B128 1
#  else
#    define HAVE_ASYNC_B128 0
#  endif
#  if __has_builtin(__builtin_amdgcn_s_wait_asynccnt)
#    define HAVE_WAIT_ASYNC 1
#  else
#    define HAVE_WAIT_ASYNC 0
#  endif
#else
#  define HAVE_ASYNC_B128 0
#  define HAVE_WAIT_ASYNC 0
#endif

// async-to-LDS builtin wants pointers to 16B int vectors in AS1 (global) / AS3 (LDS)
typedef int v4i __attribute__((vector_size(16)));
typedef __attribute__((address_space(1))) v4i gv4i;
typedef __attribute__((address_space(3))) v4i sv4i;

// ---------------------------------------------------------------- f32 -> bf16 (x4)
__global__ void k_f2bf(const float* __restrict__ src, bf16* __restrict__ dst, size_t n) {
  size_t i = ((size_t)blockIdx.x * blockDim.x + threadIdx.x) * 4;
  if (i + 3 < n) {
    float4 v = *(const float4*)(src + i);
    dst[i] = (bf16)v.x; dst[i+1] = (bf16)v.y; dst[i+2] = (bf16)v.z; dst[i+3] = (bf16)v.w;
  } else {
    for (; i < n; ++i) dst[i] = (bf16)src[i];
  }
}

// ---------------------------------------------------------------- embedding
__global__ void k_embed(const int* __restrict__ tar, const float* __restrict__ emb,
                        float* __restrict__ te, bf16* __restrict__ tebf) {
  int i = blockIdx.x * blockDim.x + threadIdx.x;     // over BT_*E_
  if (i >= BT_ * E_) return;
  int row = i / E_, e = i % E_;
  int tok = tar[row];
  float v = (tok == 0) ? 0.f : emb[(size_t)tok * E_ + e];  // padding_idx=0 row zeroed
  te[i] = v; tebf[i] = (bf16)v;
}

__global__ void k_lengths(const int* __restrict__ tar, int* __restrict__ len) {
  int b = threadIdx.x;
  if (b < B_) {
    int c = 0;
    for (int t = 0; t < T_; ++t) c += (tar[b * T_ + t] > 0) ? 1 : 0;
    len[b] = c > 0 ? c : 1;
  }
}

__global__ void k_hinit(const float* __restrict__ hidden, int layer,
                        float* __restrict__ h, bf16* __restrict__ hbf) {
  int i = blockIdx.x * blockDim.x + threadIdx.x;
  if (i < B_ * H_) {
    float v = hidden[(size_t)layer * B_ * H_ + i];
    h[i] = v; hbf[i] = (bf16)v;
  }
}

// ---------------------------------------------------------------- bf16 WMMA GEMM
// C[M,N] = A[M,K] @ W[N,K]^T + bias[N]   (A,W bf16 row-major; C f32; optional bf16 copy)
// Block = 256 threads = 8 waves. One block covers one 16-row A tile x 8 N tiles.
// The 16xK A tile is staged into LDS once (async global->LDS when available),
// zero-padded for rows >= M; all 8 waves then read fragments from LDS (ds_load).
// W tiles stream from global (re-reads hit the 192MB L2).
__global__ void k_gemm_wmma(const bf16* __restrict__ A, const bf16* __restrict__ W,
                            const float* __restrict__ bias, float* __restrict__ C,
                            bf16* __restrict__ Cbf, int M, int N, int K) {
  extern __shared__ char smem[];                       // 16 * K * 2 bytes
  const int tid  = threadIdx.x;
  const int lane = tid & 31;
  const int wib  = tid >> 5;                           // wave in block
  const int ntiles  = N >> 4;
  const int ngroups = (ntiles + 7) >> 3;
  const int tm = blockIdx.x / ngroups;                 // M tile
  const int tn = (blockIdx.x % ngroups) * 8 + wib;     // N tile for this wave
  const int rowBytes = K * 2;

  // ---- stage A tile [16 x K] into LDS (16B chunks), zero-pad invalid rows
  const int chunksPerRow = rowBytes >> 4;
  const int totalChunks  = 16 * chunksPerRow;
  for (int c = tid; c < totalChunks; c += 256) {
    int row  = c / chunksPerRow;
    int boff = (c % chunksPerRow) << 4;
    int mg   = tm * 16 + row;
    char* ldst = smem + row * rowBytes + boff;
    if (mg < M) {
      const char* gsrc = (const char*)A + (size_t)mg * rowBytes + boff;
#if HAVE_ASYNC_B128
      __builtin_amdgcn_global_load_async_to_lds_b128(
          (gv4i*)(uintptr_t)gsrc,
          (sv4i*)(unsigned)(uintptr_t)ldst, 0, 0);
#else
      *(int4*)ldst = *(const int4*)gsrc;
#endif
    } else {
      int4 z = make_int4(0, 0, 0, 0);
      *(int4*)ldst = z;
    }
  }
#if HAVE_ASYNC_B128
#  if HAVE_WAIT_ASYNC
  __builtin_amdgcn_s_wait_asynccnt(0);
#  else
  asm volatile("s_wait_asynccnt 0x0" ::: "memory");
#  endif
#endif
  __syncthreads();

  if (tn < ntiles) {
    const int half = lane >> 4;        // K-half selector per ISA fragment layout
    const int r16  = lane & 15;
    const bf16* arow = (const bf16*)(smem + r16 * rowBytes);   // LDS, row = local M
    const bf16* wrow = W + (size_t)(tn * 16 + r16) * K;        // global, row = out col
    v8f acc = {0.f, 0.f, 0.f, 0.f, 0.f, 0.f, 0.f, 0.f};
    for (int k0 = 0; k0 < K; k0 += 32) {
      __builtin_prefetch((const void*)(wrow + k0 + 128), 0, 1);   // stream weights
      v8bf alo = *(const v8bf*)(arow + k0 + 8 * half);            // K = 8h..8h+7
      v8bf ahi = *(const v8bf*)(arow + k0 + 16 + 8 * half);       // K = 16+8h..23+8h
      v8bf blo = *(const v8bf*)(wrow + k0 + 8 * half);
      v8bf bhi = *(const v8bf*)(wrow + k0 + 16 + 8 * half);
      v16bf a = __builtin_shufflevector(alo, ahi, 0,1,2,3,4,5,6,7,8,9,10,11,12,13,14,15);
      v16bf b = __builtin_shufflevector(blo, bhi, 0,1,2,3,4,5,6,7,8,9,10,11,12,13,14,15);
      acc = __builtin_amdgcn_wmma_f32_16x16x32_bf16(false, a, false, b,
                                                    (short)0, acc, false, false);
    }
    const int n = tn * 16 + r16;
    const float bn = bias ? bias[n] : 0.f;
#pragma unroll
    for (int r = 0; r < 8; ++r) {      // C/D: VGPR r -> M = r + 8*half, N = r16
      int mo = tm * 16 + r + 8 * half;
      if (mo < M) {
        float v = acc[r] + bn;
        C[(size_t)mo * N + n] = v;
        if (Cbf) Cbf[(size_t)mo * N + n] = (bf16)v;
      }
    }
  }
}

// ---------------------------------------------------------------- GRU pointwise step
__global__ void k_gru_step(const float* __restrict__ GI, const float* __restrict__ GH,
                           int t, float* __restrict__ h, bf16* __restrict__ hbf,
                           float* __restrict__ Yf, bf16* __restrict__ Ybf) {
  int i = blockIdx.x * blockDim.x + threadIdx.x;
  if (i >= B_ * H_) return;
  int b = i / H_, j = i % H_;
  size_t gi = ((size_t)(b * T_ + t)) * G3H_;
  size_t gh = (size_t)b * G3H_;
  float ir = GI[gi + j], iz = GI[gi + H_ + j], in = GI[gi + 2 * H_ + j];
  float hr = GH[gh + j], hz = GH[gh + H_ + j], hn = GH[gh + 2 * H_ + j];
  float r = 1.f / (1.f + __expf(-(ir + hr)));
  float z = 1.f / (1.f + __expf(-(iz + hz)));
  float n = tanhf(in + r * hn);
  float hv = (1.f - z) * n + z * h[i];
  h[i] = hv; hbf[i] = (bf16)hv;
  size_t y = ((size_t)(b * T_ + t)) * H_ + j;
  Yf[y] = hv; Ybf[y] = (bf16)hv;
}

// ---------------------------------------------------------------- pack_padded mask
__global__ void k_decmask(const float* __restrict__ Yf, const int* __restrict__ len,
                          float* __restrict__ dec, bf16* __restrict__ decbf) {
  int i = blockIdx.x * blockDim.x + threadIdx.x;    // over BT_*H_
  if (i >= BT_ * H_) return;
  int row = i / H_, b = row / T_, t = row % T_;
  float v = (t < len[b]) ? Yf[i] : 0.f;
  dec[i] = v; decbf[i] = (bf16)v;
}

// ---------------------------------------------------------------- additive attention
__global__ void k_attention(const float* __restrict__ e1, const float* __restrict__ e2,
                            const float* __restrict__ vw, const int* __restrict__ sou,
                            const float* __restrict__ enc, const bf16* __restrict__ decbf,
                            float* __restrict__ att_out, float* __restrict__ weighted,
                            bf16* __restrict__ allout_bf) {
  __shared__ float sm[S_];
  __shared__ float red[S_];
  const int row = blockIdx.x;        // b*T + t
  const int b = row / T_;
  const int s = threadIdx.x;         // blockDim == 256 == S_
  const float* e1r = e1 + ((size_t)(b * S_ + s)) * H_;
  const float* e2r = e2 + (size_t)row * H_;
  float acc = 0.f;
  for (int hh = 0; hh < H_; ++hh) acc += vw[hh] * tanhf(e1r[hh] + e2r[hh]);
  if (sou[b * S_ + s] == 0) acc = -INFINITY;
  red[s] = acc; __syncthreads();
  for (int o = 128; o > 0; o >>= 1) { if (s < o) red[s] = fmaxf(red[s], red[s + o]); __syncthreads(); }
  float mx = red[0]; __syncthreads();
  float p = __expf(acc - mx);
  red[s] = p; __syncthreads();
  for (int o = 128; o > 0; o >>= 1) { if (s < o) red[s] += red[s + o]; __syncthreads(); }
  float a = p / red[0];
  sm[s] = a;
  att_out[(size_t)row * S_ + s] = a;
  __syncthreads();
  for (int hh = s; hh < H_; hh += S_) {
    float w = 0.f;
    for (int s2 = 0; s2 < S_; ++s2) w += sm[s2] * enc[((size_t)(b * S_ + s2)) * H_ + hh];
    weighted[(size_t)row * H_ + hh] = w;
    allout_bf[(size_t)row * 2 * H_ + hh]      = (bf16)w;                       // concat(weighted,
    allout_bf[(size_t)row * 2 * H_ + H_ + hh] = decbf[(size_t)row * H_ + hh];  //        dec_out)
  }
}

// ---------------------------------------------------------------- copy gate
__global__ void k_gate(const float* __restrict__ weighted, const float* __restrict__ dec,
                       const float* __restrict__ te,
                       const float* __restrict__ g1w, const float* __restrict__ g1b,
                       const float* __restrict__ g2w, const float* __restrict__ g2b,
                       const float* __restrict__ g3w, const float* __restrict__ g3b,
                       float* __restrict__ gate) {
  __shared__ float red[256];
  int row = blockIdx.x, tid = threadIdx.x;
  float acc = 0.f;
  for (int hh = tid; hh < H_; hh += 256)
    acc += weighted[(size_t)row * H_ + hh] * g1w[hh]
         + dec[(size_t)row * H_ + hh] * g2w[hh]
         + te[(size_t)row * E_ + hh] * g3w[hh];
  red[tid] = acc; __syncthreads();
  for (int o = 128; o > 0; o >>= 1) { if (tid < o) red[tid] += red[tid + o]; __syncthreads(); }
  if (tid == 0) gate[row] = 1.f / (1.f + __expf(-(red[0] + g1b[0] + g2b[0] + g3b[0])));
}

// ------------------------------------------- masked softmax + scatter copy + blend (in place)
__global__ void k_final(float* __restrict__ out, const float* __restrict__ att,
                        const float* __restrict__ gate, const int* __restrict__ ext_len,
                        const int* __restrict__ sou) {
  __shared__ float red[256];
  int row = blockIdx.x, tid = threadIdx.x;
  int b = row / T_;
  int limit = VOCAB_ + ext_len[b];
  float* orow = out + (size_t)row * VEXT_;
  float mx = -INFINITY;
  for (int c = tid; c < VEXT_; c += 256) if (c < limit) mx = fmaxf(mx, orow[c]);
  red[tid] = mx; __syncthreads();
  for (int o = 128; o > 0; o >>= 1) { if (tid < o) red[tid] = fmaxf(red[tid], red[tid + o]); __syncthreads(); }
  mx = red[0]; __syncthreads();
  float sum = 0.f;
  for (int c = tid; c < VEXT_; c += 256) if (c < limit) sum += __expf(orow[c] - mx);
  red[tid] = sum; __syncthreads();
  for (int o = 128; o > 0; o >>= 1) { if (tid < o) red[tid] += red[tid + o]; __syncthreads(); }
  float inv = 1.f / red[0]; __syncthreads();
  float g = gate[row], og = 1.f - g;
  for (int c = tid; c < VEXT_; c += 256)
    orow[c] = (c < limit) ? og * __expf(orow[c] - mx) * inv : 0.f;
  __syncthreads();
  // copy mechanism: scatter-add gate * attention into extended vocab
  if (tid < S_) {
    int col = sou[b * S_ + tid];
    atomicAdd(&orow[col], g * att[(size_t)row * S_ + tid]);
  }
}

// ================================================================ host
static inline unsigned gemm_blocks(int M, int N) {
  int mtiles  = (M + 15) / 16;
  int ngroups = ((N / 16) + 7) / 8;
  return (unsigned)(mtiles * ngroups);
}

extern "C" void kernel_launch(void* const* d_in, const int* in_sizes, int n_in,
                              void* d_out, int out_size, void* d_ws, size_t ws_size,
                              hipStream_t stream) {
  (void)in_sizes; (void)n_in; (void)out_size;
  const int*   sou      = (const int*)  d_in[0];
  const int*   tar      = (const int*)  d_in[1];
  const float* hidden   = (const float*)d_in[2];
  const float* enc      = (const float*)d_in[3];
  const int*   ext_len  = (const int*)  d_in[4];
  const float* emb      = (const float*)d_in[5];
  const float* gru_wih  = (const float*)d_in[6];
  const float* gru_whh  = (const float*)d_in[7];
  const float* gru_bih  = (const float*)d_in[8];
  const float* gru_bhh  = (const float*)d_in[9];
  const float* fc_enc_w = (const float*)d_in[10];
  const float* fc_enc_b = (const float*)d_in[11];
  const float* fc_dec_w = (const float*)d_in[12];
  const float* fc_dec_b = (const float*)d_in[13];
  const float* v_w      = (const float*)d_in[14];
  const float* fc_out1_w= (const float*)d_in[15];
  const float* fc_out1_b= (const float*)d_in[16];
  const float* fc_out2_w= (const float*)d_in[17];
  const float* fc_out2_b= (const float*)d_in[18];
  const float* g1w = (const float*)d_in[19]; const float* g1b = (const float*)d_in[20];
  const float* g2w = (const float*)d_in[21]; const float* g2b = (const float*)d_in[22];
  const float* g3w = (const float*)d_in[23]; const float* g3b = (const float*)d_in[24];

  float* out_logits = (float*)d_out;                       // [BT, VEXT], softmaxed in place
  float* out_att    = (float*)d_out + (size_t)BT_ * VEXT_; // [BT, S]

  // -------- carve workspace
  unsigned char* p = (unsigned char*)d_ws;
  auto carve = [&](size_t bytes) -> void* {
    void* r = (void*)p; p += (bytes + 255) & ~(size_t)255; return r;
  };
  const size_t WIH_N = (size_t)L_ * G3H_ * H_;
  bf16*  wih_bf   = (bf16*) carve(WIH_N * 2);
  bf16*  whh_bf   = (bf16*) carve(WIH_N * 2);
  bf16*  fcenc_bf = (bf16*) carve((size_t)H_ * H_ * 2);
  bf16*  fcdec_bf = (bf16*) carve((size_t)H_ * H_ * 2);
  bf16*  fcout2_bf= (bf16*) carve((size_t)H_ * 2 * H_ * 2);
  bf16*  fcout1_bf= (bf16*) carve((size_t)VEXT_ * H_ * 2);
  bf16*  enc_bf   = (bf16*) carve((size_t)B_ * S_ * H_ * 2);
  float* te       = (float*)carve((size_t)BT_ * E_ * 4);
  bf16*  te_bf    = (bf16*) carve((size_t)BT_ * E_ * 2);
  int*   len      = (int*)  carve(B_ * 4);
  float* h        = (float*)carve((size_t)B_ * H_ * 4);
  bf16*  h_bf     = (bf16*) carve((size_t)B_ * H_ * 2);
  float* GI       = (float*)carve((size_t)BT_ * G3H_ * 4);
  float* GH       = (float*)carve((size_t)B_ * G3H_ * 4);
  float* Yf       = (float*)carve((size_t)BT_ * H_ * 4);
  bf16*  Ybf      = (bf16*) carve((size_t)BT_ * H_ * 2);
  float* dec      = (float*)carve((size_t)BT_ * H_ * 4);
  bf16*  dec_bf   = (bf16*) carve((size_t)BT_ * H_ * 2);
  float* e1       = (float*)carve((size_t)B_ * S_ * H_ * 4);
  float* e2       = (float*)carve((size_t)BT_ * H_ * 4);
  float* weighted = (float*)carve((size_t)BT_ * H_ * 4);
  bf16*  allout_bf= (bf16*) carve((size_t)BT_ * 2 * H_ * 2);
  float* mid      = (float*)carve((size_t)BT_ * H_ * 4);
  bf16*  mid_bf   = (bf16*) carve((size_t)BT_ * H_ * 2);
  float* gate     = (float*)carve((size_t)BT_ * 4);
  if ((size_t)(p - (unsigned char*)d_ws) > ws_size) return;  // insufficient scratch

  // -------- weight / activation conversions to bf16
  auto f2bf = [&](const float* s, bf16* d, size_t n) {
    k_f2bf<<<dim3((unsigned)((n + 1023) / 1024)), 256, 0, stream>>>(s, d, n);
  };
  f2bf(gru_wih, wih_bf, WIH_N);
  f2bf(gru_whh, whh_bf, WIH_N);
  f2bf(fc_enc_w, fcenc_bf, (size_t)H_ * H_);
  f2bf(fc_dec_w, fcdec_bf, (size_t)H_ * H_);
  f2bf(fc_out2_w, fcout2_bf, (size_t)H_ * 2 * H_);
  f2bf(fc_out1_w, fcout1_bf, (size_t)VEXT_ * H_);
  f2bf(enc, enc_bf, (size_t)B_ * S_ * H_);

  k_embed<<<dim3((BT_ * E_ + 255) / 256), 256, 0, stream>>>(tar, emb, te, te_bf);
  k_lengths<<<dim3(1), 32, 0, stream>>>(tar, len);

  auto gemm = [&](const bf16* A, const bf16* W, const float* bias, float* C, bf16* Cbf,
                  int M, int N, int K) {
    size_t sh = (size_t)16 * K * 2;
    k_gemm_wmma<<<dim3(gemm_blocks(M, N)), 256, sh, stream>>>(A, W, bias, C, Cbf, M, N, K);
  };

  // -------- 4-layer GRU: hoisted input projection + sequential recurrence
  const size_t WSTRIDE = (size_t)G3H_ * H_;
  for (int l = 0; l < L_; ++l) {
    k_hinit<<<dim3((B_ * H_ + 255) / 256), 256, 0, stream>>>(hidden, l, h, h_bf);
    const bf16* X = (l == 0) ? te_bf : Ybf;
    gemm(X, wih_bf + (size_t)l * WSTRIDE, gru_bih + (size_t)l * G3H_,
         GI, (bf16*)nullptr, BT_, G3H_, H_);
    for (int t = 0; t < T_; ++t) {
      gemm(h_bf, whh_bf + (size_t)l * WSTRIDE, gru_bhh + (size_t)l * G3H_,
           GH, (bf16*)nullptr, B_, G3H_, H_);
      k_gru_step<<<dim3((B_ * H_ + 255) / 256), 256, 0, stream>>>(
          GI, GH, t, h, h_bf, Yf, Ybf);
    }
  }
  k_decmask<<<dim3((BT_ * H_ + 255) / 256), 256, 0, stream>>>(Yf, len, dec, dec_bf);

  // -------- attention projections (WMMA) + energy/softmax/context
  gemm(enc_bf, fcenc_bf, fc_enc_b, e1, (bf16*)nullptr, B_ * S_, H_, H_);
  gemm(dec_bf, fcdec_bf, fc_dec_b, e2, (bf16*)nullptr, BT_, H_, H_);
  k_attention<<<dim3(BT_), S_, 0, stream>>>(
      e1, e2, v_w, sou, enc, dec_bf, out_att, weighted, allout_bf);

  k_gate<<<dim3(BT_), 256, 0, stream>>>(weighted, dec, te,
                                        g1w, g1b, g2w, g2b, g3w, g3b, gate);

  // -------- output projections: fc_out2 then the dominant fc_out1 (logits -> d_out)
  gemm(allout_bf, fcout2_bf, fc_out2_b, mid, mid_bf, BT_, H_, 2 * H_);
  gemm(mid_bf, fcout1_bf, fc_out1_b, out_logits, (bf16*)nullptr, BT_, VEXT_, H_);

  // -------- masked softmax + copy-mechanism scatter + gate blend (in place)
  k_final<<<dim3(BT_), 256, 0, stream>>>(out_logits, out_att, gate, ext_len, sou);
}
